// QKVSeperateExpert_89189290869040
// MI455X (gfx1250) — matrix operand order repaired and checked by reading, not compiled, not run on hardware
//
#include <hip/hip_runtime.h>

// MoE grouped GEMM: out[T,N] = inp[T,K] @ W_e[N,K]^T + b_e, tokens contiguous
// per expert. fp32 in/out; compute via bf16x3-split WMMA (memory-bound at
// ~19us on 23.3 TB/s; bf16 WMMA compute is <1us, fp32 WMMA would be ~100us).

#define E_NUM 16
#define K_DIM 1024
#define N_DIM 3072
#define T_TOT 16384

#define BM 128
#define BN 128
#define BK 32

typedef __attribute__((ext_vector_type(16))) __bf16 v16bf;
typedef __attribute__((ext_vector_type(8)))  __bf16 v8bf;
typedef __attribute__((ext_vector_type(4)))  __bf16 v4bf;
typedef __attribute__((ext_vector_type(8)))  float  v8f;
typedef __attribute__((ext_vector_type(4)))  float  v4f;

__device__ __forceinline__ v16bf cat8(v8bf a, v8bf b) {
  return __builtin_shufflevector(a, b, 0,1,2,3,4,5,6,7,8,9,10,11,12,13,14,15);
}

__global__ __launch_bounds__(256)
void moe_grouped_gemm_bf16x3(const float* __restrict__ inp,
                             const int*   __restrict__ counts,
                             const float* __restrict__ wgt,
                             const float* __restrict__ bias,
                             float*       __restrict__ out)
{
  const int e    = blockIdx.z;
  const int n0   = blockIdx.x * BN;

  // Expert token range via prefix sum of the 16-entry count array.
  int start = 0;
  #pragma unroll
  for (int i = 0; i < E_NUM; ++i) start += (i < e) ? counts[i] : 0;
  const int tokens = counts[e];
  const int mblk   = blockIdx.y * BM;
  if (mblk >= tokens) return;                 // over-launched block: retire
  const int m_base     = start + mblk;
  const int rem        = tokens - mblk;
  const int rows_valid = (rem < BM) ? rem : BM;

  // LDS staging: fp32 tiles split into bf16 hi/lo. 4 * 8KB = 32KB.
  __shared__ __bf16 Ahi[BM * BK];
  __shared__ __bf16 Alo[BM * BK];
  __shared__ __bf16 Bhi[BN * BK];
  __shared__ __bf16 Blo[BN * BK];

  const int tid  = threadIdx.x;
  const int lane = tid & 31;
  const int wave = tid >> 5;
  const int wm   = (wave & 1) * 64;   // wave's M offset within 128-row tile
  const int wn   = (wave >> 1) * 32;  // wave's N offset within 128-col tile
  const int lrow = lane & 15;
  const int lhi  = lane >> 4;         // half-wave select (ISA frag layouts)

  v8f acc[4][2] = {};                 // 4 M-tiles x 2 N-tiles of 16x16 f32

  const float* wb = wgt + ((size_t)e * N_DIM + n0) * K_DIM;

  #pragma unroll 1
  for (int k0 = 0; k0 < K_DIM; k0 += BK) {
    __syncthreads();  // previous iteration's fragments consumed

    // ---- Stage A (inp) and B (weight) tiles, fp32 -> bf16 hi/lo split ----
    // 128 rows x 32 cols fp32 per tile; 256 threads x 4 iters x float4.
    #pragma unroll
    for (int it = 0; it < 4; ++it) {
      const int row = (tid >> 3) + it * 32;   // 0..127
      const int c4  = (tid & 7) * 4;          // 0,4,...,28

      // A tile: rows past this expert's range are garbage-but-unread at store
      v4f av = {0.f, 0.f, 0.f, 0.f};
      const int gr = m_base + row;
      if (gr < T_TOT)
        av = *(const v4f*)(inp + (size_t)gr * K_DIM + k0 + c4);
      v4bf ah, al;
      #pragma unroll
      for (int j = 0; j < 4; ++j) {
        const float x = av[j];
        const __bf16 h = (__bf16)x;           // RNE truncate to bf16
        ah[j] = h;
        al[j] = (__bf16)(x - (float)h);       // residual
      }
      *(v4bf*)&Ahi[row * BK + c4] = ah;
      *(v4bf*)&Alo[row * BK + c4] = al;

      // B tile: weight[e][n0+row][k0+c4..+3], contiguous in K
      const v4f bv = *(const v4f*)(wb + (size_t)row * K_DIM + k0 + c4);
      v4bf bh, bl;
      #pragma unroll
      for (int j = 0; j < 4; ++j) {
        const float x = bv[j];
        const __bf16 h = (__bf16)x;
        bh[j] = h;
        bl[j] = (__bf16)(x - (float)h);
      }
      *(v4bf*)&Bhi[row * BK + c4] = bh;
      *(v4bf*)&Blo[row * BK + c4] = bl;
    }
    __syncthreads();

    // ---- Load WMMA fragments in contiguous k order (A and B match) ----
    // A 16x32 bf16: lane<16 -> row=lane, elems = K0..15;
    //               lane>=16 -> row=lane-16, elems = K16..31.
    v16bf afh[4], afl[4];
    #pragma unroll
    for (int mt = 0; mt < 4; ++mt) {
      const int r = (wm + mt * 16 + lrow) * BK + lhi * 16;
      afh[mt] = *(const v16bf*)&Ahi[r];
      afl[mt] = *(const v16bf*)&Alo[r];
    }
    // B 32x16 bf16: lane<16 -> col=lane, elems = K0..15; lane>=16 -> K16..31.
    v16bf bfh[2], bfl[2];
    #pragma unroll
    for (int nt = 0; nt < 2; ++nt) {
      const int r = (wn + nt * 16 + lrow) * BK + lhi * 16;
      bfh[nt] = *(const v16bf*)&Bhi[r];
      bfl[nt] = *(const v16bf*)&Blo[r];
    }

    // ---- 24 WMMAs: bf16x3 split (hi*hi + hi*lo + lo*hi) ----
    #pragma unroll
    for (int mt = 0; mt < 4; ++mt) {
      #pragma unroll
      for (int nt = 0; nt < 2; ++nt) {
        acc[mt][nt] = __builtin_amdgcn_wmma_f32_16x16x32_bf16(
            false, afh[mt], false, bfh[nt], (short)0, acc[mt][nt], false, false);
        acc[mt][nt] = __builtin_amdgcn_wmma_f32_16x16x32_bf16(
            false, afh[mt], false, bfl[nt], (short)0, acc[mt][nt], false, false);
        acc[mt][nt] = __builtin_amdgcn_wmma_f32_16x16x32_bf16(
            false, afl[mt], false, bfh[nt], (short)0, acc[mt][nt], false, false);
      }
    }
  }

  // ---- Epilogue: bias add + bounded store ----
  // C 16x16 f32 layout: lane<16 -> N=lane, M=r; lane>=16 -> N=lane-16, M=r+8.
  #pragma unroll
  for (int nt = 0; nt < 2; ++nt) {
    const int   gcol = n0 + wn + nt * 16 + lrow;
    const float bv   = bias[(size_t)e * N_DIM + gcol];
    #pragma unroll
    for (int mt = 0; mt < 4; ++mt) {
      #pragma unroll
      for (int r = 0; r < 8; ++r) {
        const int lr = wm + mt * 16 + lhi * 8 + r;
        if (lr < rows_valid)
          out[(size_t)(m_base + lr) * N_DIM + gcol] = acc[mt][nt][r] + bv;
      }
    }
  }
}

extern "C" void kernel_launch(void* const* d_in, const int* in_sizes, int n_in,
                              void* d_out, int out_size, void* d_ws, size_t ws_size,
                              hipStream_t stream) {
  (void)in_sizes; (void)n_in; (void)out_size; (void)d_ws; (void)ws_size;
  const float* inp  = (const float*)d_in[0];
  const int*   cnt  = (const int*)d_in[1];   // fwd_expert_count
  const float* wgt  = (const float*)d_in[2];
  const float* bias = (const float*)d_in[3];
  float*       out  = (float*)d_out;

  // grid.y sized for the worst case (one expert owns all tokens); blocks
  // outside their expert's token range retire after the scalar count read.
  dim3 grid(N_DIM / BN, (T_TOT + BM - 1) / BM, E_NUM);
  dim3 block(256);
  hipLaunchKernelGGL(moe_grouped_gemm_bf16x3, grid, block, 0, stream,
                     inp, cnt, wgt, bias, out);
}
